// VectorQuantizer_30056181137415
// MI455X (gfx1250) — matrix operand order repaired and checked by reading, not compiled, not run on hardware
//
#include <hip/hip_runtime.h>

typedef __attribute__((ext_vector_type(16))) _Float16 v16h;
typedef __attribute__((ext_vector_type(8)))  _Float16 v8h;
typedef __attribute__((ext_vector_type(8)))  float    v8f;
typedef __attribute__((ext_vector_type(2)))  float    v2f;

#define N_TOK 16384
#define N_E   8192
#define E_DIM 256
#define NSPLIT 4
#define CODES_PER_SPLIT (N_E / NSPLIT)   // 2048

// ---------------- workspace layout (bytes) ----------------
#define WS_ZH    0                      // N_TOK*E_DIM*2 = 8,388,608
#define WS_EMBH  8388608                // N_E*E_DIM*2   = 4,194,304
#define WS_ENORM 12582912               // N_E*4         = 32,768
#define WS_IDX   12615680               // N_TOK*4       = 65,536
#define WS_HIST  12681216               // N_E*4         = 32,768
#define WS_ACC   12713984               // 4
#define WS_PMIN  12714240               // N_TOK*NSPLIT*4 = 262,144
#define WS_PIDX  12976384               // N_TOK*NSPLIT*4 = 262,144
// total ~13.2 MB

// K0a: convert z -> f16, zero histogram + loss accumulator
__global__ void k_convert_z(const float* __restrict__ z, _Float16* __restrict__ zh,
                            int* __restrict__ hist, float* __restrict__ accum) {
    int i = blockIdx.x * 256 + threadIdx.x;
    if (i < N_TOK * E_DIM) zh[i] = (_Float16)z[i];
    if (i < N_E) hist[i] = 0;
    if (i == 0) *accum = 0.0f;
}

// K0b: convert emb -> f16 and compute ||e_c||^2 (one block of 256 per code row)
__global__ void k_convert_emb(const float* __restrict__ emb, _Float16* __restrict__ eh,
                              float* __restrict__ enorm) {
    __shared__ float red[8];
    int c = blockIdx.x;
    int d = threadIdx.x;
    float e = emb[(size_t)c * E_DIM + d];
    eh[(size_t)c * E_DIM + d] = (_Float16)e;
    float s = e * e;
    #pragma unroll
    for (int o = 16; o > 0; o >>= 1) s += __shfl_xor(s, o, 32);
    if ((threadIdx.x & 31) == 0) red[threadIdx.x >> 5] = s;
    __syncthreads();
    if (threadIdx.x == 0) {
        float t = 0.f;
        #pragma unroll
        for (int i = 0; i < 8; ++i) t += red[i];
        enorm[c] = t;
    }
}

// Load one B tile (Kx16 codes, f16 WMMA layout): lane l16 = code column,
// lane half selects K 0..15 vs 16..31 (contiguous 32B per lane).
__device__ __forceinline__ void load_btile(const _Float16* __restrict__ eh,
                                           int c, int l16, int half, v16h B[8]) {
    const _Float16* p = eh + (size_t)(c + l16) * E_DIM + half * 16;
    #pragma unroll
    for (int kb = 0; kb < 8; ++kb) B[kb] = *(const v16h*)(p + kb * 32);
}

__device__ __forceinline__ void minupd(const v8f& acc, float en, int code,
                                       float rmin[8], int ridx[8]) {
    #pragma unroll
    for (int r = 0; r < 8; ++r) {
        float s = fmaf(-2.0f, acc[r], en);   // ||e||^2 - 2 z.e
        bool lt = (s < rmin[r]);
        rmin[r] = lt ? s : rmin[r];
        ridx[r] = lt ? code : ridx[r];
    }
}

// K1: fused WMMA GEMM + running argmin, 4-way code split, B double-buffered.
// blockIdx & 3 = code split; 8 waves/block each own one 16-token tile.
__global__ void __launch_bounds__(256) k_argmin(const _Float16* __restrict__ zh,
                                                const _Float16* __restrict__ eh,
                                                const float* __restrict__ enorm,
                                                float* __restrict__ pmin,
                                                int* __restrict__ pidx) {
    const int wave = threadIdx.x >> 5;
    const int lane = threadIdx.x & 31;
    const int half = lane >> 4;
    const int l16  = lane & 15;
    const int cs   = blockIdx.x & (NSPLIT - 1);
    const int t0   = ((blockIdx.x >> 2) * 8 + wave) * 16;
    const int cBeg = cs * CODES_PER_SPLIT;
    const int cEnd = cBeg + CODES_PER_SPLIT;

    // ---- A tile resident: 16 tokens x 256 dims (64 VGPRs) ----
    // lane<16 : VGPR0-3 = K k0+0..7,   VGPR4-7 = K k0+8..15
    // lane>=16: VGPR0-3 = K k0+16..23, VGPR4-7 = K k0+24..31
    v16h A[8];
    const _Float16* zrow = zh + (size_t)(t0 + l16) * E_DIM;
    #pragma unroll
    for (int kb = 0; kb < 8; ++kb) {
        const int k0 = kb * 32;
        *((v8h*)&A[kb])       = *(const v8h*)(zrow + k0 + half * 16);
        *(((v8h*)&A[kb]) + 1) = *(const v8h*)(zrow + k0 + half * 16 + 8);
    }

    float rmin[8];
    int   ridx[8];
    #pragma unroll
    for (int r = 0; r < 8; ++r) { rmin[r] = 3.0e38f; ridx[r] = 0; }

    // ---- ping-pong double buffer over code tiles (128 tiles, 64 iters) ----
    v16h Bb[8], Bn[8];
    load_btile(eh, cBeg, l16, half, Bb);
    for (int c0 = cBeg; c0 < cEnd; c0 += 32) {
        load_btile(eh, c0 + 16, l16, half, Bn);   // always in range (even tiles)
        {
            v8f acc = {};
            #pragma unroll
            for (int kb = 0; kb < 8; ++kb)
                acc = __builtin_amdgcn_wmma_f32_16x16x32_f16(
                          false, A[kb], false, Bb[kb], (short)0, acc, false, false);
            minupd(acc, enorm[c0 + l16], c0 + l16, rmin, ridx);
        }
        int c2 = (c0 + 32 < cEnd) ? (c0 + 32) : cBeg;   // wrap prefetch (harmless)
        load_btile(eh, c2, l16, half, Bb);
        {
            v8f acc = {};
            #pragma unroll
            for (int kb = 0; kb < 8; ++kb)
                acc = __builtin_amdgcn_wmma_f32_16x16x32_f16(
                          false, A[kb], false, Bn[kb], (short)0, acc, false, false);
            minupd(acc, enorm[c0 + 16 + l16], c0 + 16 + l16, rmin, ridx);
        }
    }

    // ---- argmin across the 16 lanes holding one token's code columns ----
    #pragma unroll
    for (int r = 0; r < 8; ++r) {
        #pragma unroll
        for (int o = 1; o < 16; o <<= 1) {
            float om = __shfl_xor(rmin[r], o, 32);
            int   oi = __shfl_xor(ridx[r], o, 32);
            if (om < rmin[r] || (om == rmin[r] && oi < ridx[r])) {
                rmin[r] = om; ridx[r] = oi;
            }
        }
        if (l16 == 0) {
            int t = t0 + half * 8 + r;
            pmin[t * NSPLIT + cs] = rmin[r];
            pidx[t * NSPLIT + cs] = ridx[r];
        }
    }
}

// K1b: combine the NSPLIT partial argmins (splits are in increasing code order,
// so strict < keeps first-occurrence semantics of jnp.argmin)
__global__ void k_reduce(const float* __restrict__ pmin, const int* __restrict__ pidx,
                         int* __restrict__ idx, float* __restrict__ idxout) {
    int t = blockIdx.x * 256 + threadIdx.x;
    if (t < N_TOK) {
        float m = pmin[t * NSPLIT];
        int   ix = pidx[t * NSPLIT];
        #pragma unroll
        for (int s = 1; s < NSPLIT; ++s) {
            float om = pmin[t * NSPLIT + s];
            int   oi = pidx[t * NSPLIT + s];
            if (om < m) { m = om; ix = oi; }
        }
        idx[t] = ix;
        idxout[t] = (float)ix;
    }
}

// K2: gather z_q rows (f32 from original emb), loss partial sum, histogram
__global__ void k_gather(const float* __restrict__ z, const float* __restrict__ emb,
                         const int* __restrict__ idx, float* __restrict__ zq_out,
                         int* __restrict__ hist, float* __restrict__ accum) {
    __shared__ float red[8];
    int t = blockIdx.x;
    int c = idx[t];
    int d = threadIdx.x;
    float e  = emb[(size_t)c * E_DIM + d];
    float zv = z[(size_t)t * E_DIM + d];
    zq_out[(size_t)t * E_DIM + d] = e;       // z_q_st == z + (z_q - z) == z_q
    float diff = e - zv;
    float s = diff * diff;
    #pragma unroll
    for (int o = 16; o > 0; o >>= 1) s += __shfl_xor(s, o, 32);
    if ((threadIdx.x & 31) == 0) red[threadIdx.x >> 5] = s;
    __syncthreads();
    if (threadIdx.x == 0) {
        float tt = 0.f;
        #pragma unroll
        for (int i = 0; i < 8; ++i) tt += red[i];
        atomicAdd(accum, tt);
        atomicAdd(hist + c, 1);
    }
}

// K3: one-hot rows in a single pass (dominant 512MB output; float2 stores,
// base is only 8-byte aligned in the flat output)
__global__ void k_onehot(const int* __restrict__ idx, float* __restrict__ enc) {
    int t = blockIdx.x;
    int c = idx[t];
    v2f* row = (v2f*)(enc + (size_t)t * N_E);
    int jc = c >> 1;
    v2f one0 = {1.0f, 0.0f}, one1 = {0.0f, 1.0f}, zero = {0.0f, 0.0f};
    v2f onev = (c & 1) ? one1 : one0;
    #pragma unroll
    for (int it = 0; it < (N_E / 2) / 256; ++it) {
        int j = it * 256 + threadIdx.x;
        row[j] = (j == jc) ? onev : zero;
    }
}

// K4: perplexity from histogram + finalize loss
__global__ void k_final(const int* __restrict__ hist, const float* __restrict__ accum,
                        float* __restrict__ loss, float* __restrict__ perp) {
    __shared__ float red[32];
    int tid = threadIdx.x;           // 1024 threads
    float s = 0.f;
    for (int i = tid; i < N_E; i += 1024) {
        float p = (float)hist[i] * (1.0f / (float)N_TOK);
        s += p * __logf(p + 1e-10f);
    }
    #pragma unroll
    for (int o = 16; o > 0; o >>= 1) s += __shfl_xor(s, o, 32);
    if ((tid & 31) == 0) red[tid >> 5] = s;
    __syncthreads();
    if (tid == 0) {
        float tot = 0.f;
        #pragma unroll
        for (int i = 0; i < 32; ++i) tot += red[i];
        *perp = __expf(-tot);
        *loss = (*accum) * ((1.0f + 0.25f) / (float)((size_t)N_TOK * E_DIM));
    }
}

extern "C" void kernel_launch(void* const* d_in, const int* in_sizes, int n_in,
                              void* d_out, int out_size, void* d_ws, size_t ws_size,
                              hipStream_t stream) {
    const float* z   = (const float*)d_in[0];
    const float* emb = (const float*)d_in[1];
    (void)in_sizes; (void)n_in; (void)out_size; (void)ws_size;

    char* ws = (char*)d_ws;
    _Float16* zh    = (_Float16*)(ws + WS_ZH);
    _Float16* embh  = (_Float16*)(ws + WS_EMBH);
    float*    enorm = (float*)(ws + WS_ENORM);
    int*      idx   = (int*)(ws + WS_IDX);
    int*      hist  = (int*)(ws + WS_HIST);
    float*    accum = (float*)(ws + WS_ACC);
    float*    pmin  = (float*)(ws + WS_PMIN);
    int*      pidx  = (int*)(ws + WS_PIDX);

    float* out    = (float*)d_out;
    float* o_loss = out;                                   // 1
    float* o_zq   = out + 1;                               // 4,194,304
    float* o_perp = out + 1 + (size_t)N_TOK * E_DIM;       // 1
    float* o_enc  = out + 2 + (size_t)N_TOK * E_DIM;       // 134,217,728
    float* o_idx  = o_enc + (size_t)N_TOK * N_E;           // 16,384

    // Phase 0: precision conversion + norms + counter init
    k_convert_z<<<(N_TOK * E_DIM + 255) / 256, 256, 0, stream>>>(z, zh, hist, accum);
    k_convert_emb<<<N_E, 256, 0, stream>>>(emb, embh, enorm);

    // Phase 1: fused WMMA distance GEMM + argmin
    // 512 blocks = 128 token-tile groups x 4 code splits; 8 waves/block
    k_argmin<<<(N_TOK / (8 * 16)) * NSPLIT, 256, 0, stream>>>(zh, embh, enorm, pmin, pidx);
    k_reduce<<<(N_TOK + 255) / 256, 256, 0, stream>>>(pmin, pidx, idx, o_idx);

    // Phase 2: gather / loss partials / histogram
    k_gather<<<N_TOK, 256, 0, stream>>>(z, emb, idx, o_zq, hist, accum);

    // Phase 3: one-hot output, single pass
    k_onehot<<<N_TOK, 256, 0, stream>>>(idx, o_enc);

    // Phase 4: scalars
    k_final<<<1, 1024, 0, stream>>>(hist, accum, o_loss, o_perp);
}
